// TreeCRFLayer_11184094838934
// MI455X (gfx1250) — compile-verified, matrix operand order checked by reading, not run on hardware
//
#include <hip/hip_runtime.h>
#include <hip/hip_bf16.h>

// TreeCRF (CKY inside/outside) for MI455X / gfx1250, wave32.
//   k_stats_*   : per-batch mean / inv-std (deterministic 2-stage reduce)
//   k_pot       : WMMA-based label-logsumexp -> 3 span potentials + E[x]
//                 (v_wmma_f32_16x16x32_f16 with all-ones B = 16 row sums/op)
//   k_inside    : CKY inside DP; (start,width) chart in 257KB padded LDS
//                 (320KB WGP LDS), (end,width) chart L2-resident in global.
//   k_outside   : outside DP + fused entropy reduction Sum(marg * E[x]).
//   k_final     : assemble (log_prob, log_prob_smooth, entropy) -> d_out[48].

#define Bn 16
#define Nn 256
#define Ln 32
#define NEGF (-1.0e9f)
#define STR (Nn + 1)   // padded LDS row stride (floats) to avoid bank conflicts

typedef __attribute__((ext_vector_type(16))) _Float16 v16h;
typedef __attribute__((ext_vector_type(8)))  float    v8f;

// ---------------- wave32 helpers ----------------

// merge two online-logsumexp states across lanes (xor pattern)
__device__ __forceinline__ void lse_merge(float& m, float& s, int mask) {
  float m2 = __shfl_xor(m, mask, 32);
  float s2 = __shfl_xor(s, mask, 32);
  float mn = fmaxf(m, m2);
  s = s * __expf(m - mn) + s2 * __expf(m2 - mn);
  m = mn;
}

__device__ __forceinline__ float pick8(v8f c, int k) {
  float r = (float)c[0];
  r = (k == 1) ? (float)c[1] : r;
  r = (k == 2) ? (float)c[2] : r;
  r = (k == 3) ? (float)c[3] : r;
  r = (k == 4) ? (float)c[4] : r;
  r = (k == 5) ? (float)c[5] : r;
  r = (k == 6) ? (float)c[6] : r;
  r = (k == 7) ? (float)c[7] : r;
  return r;
}

// WMMA f32 16x16 D layout: lanes 0-15 VGPR r = row r, lanes 16-31 VGPR r = row r+8.
// With B == ones every column is identical, so each lane's c[k] is a row sum.
// Route row-sum(row=lane) to lane (for lanes 0..15).
__device__ __forceinline__ float gather_row(v8f c, int lane) {
  float sel = pick8(c, lane & 7);
  int src = (lane & 8) ? (lane + 8) : lane;   // rows 8..15 live in the upper half
  return __shfl(sel, src & 31, 32);
}

// ---------------- stats ----------------

__global__ void k_stats_partial(const float* __restrict__ lp,
                                float* __restrict__ partS, float* __restrict__ partQ) {
  int b = blockIdx.x >> 6;
  int c = blockIdx.x & 63;
  const size_t Mb = (size_t)Nn * Nn * Ln;   // 2,097,152 per batch
  const int CH = (int)(Mb >> 6);            // 32768 per chunk
  size_t base = (size_t)b * Mb + (size_t)c * CH;
  float s = 0.f, q = 0.f;
  for (int t = threadIdx.x; t < CH; t += 256) {
    float v = lp[base + t];
    s += v;
    q = fmaf(v, v, q);
  }
  __shared__ float rs[256];
  __shared__ float rq[256];
  rs[threadIdx.x] = s; rq[threadIdx.x] = q;
  __syncthreads();
  for (int off = 128; off; off >>= 1) {
    if (threadIdx.x < off) { rs[threadIdx.x] += rs[threadIdx.x + off]; rq[threadIdx.x] += rq[threadIdx.x + off]; }
    __syncthreads();
  }
  if (threadIdx.x == 0) { partS[blockIdx.x] = rs[0]; partQ[blockIdx.x] = rq[0]; }
}

__global__ void k_stats_final(const float* __restrict__ partS, const float* __restrict__ partQ,
                              float* __restrict__ meanA, float* __restrict__ istdA) {
  __shared__ float rs[64];
  __shared__ float rq[64];
  int t = threadIdx.x, b = blockIdx.x;
  rs[t] = partS[b * 64 + t];
  rq[t] = partQ[b * 64 + t];
  __syncthreads();
  for (int off = 32; off; off >>= 1) {
    if (t < off) { rs[t] += rs[t + off]; rq[t] += rq[t + off]; }
    __syncthreads();
  }
  if (t == 0) {
    const float Mf = (float)((size_t)Nn * Nn * Ln);
    float mu = rs[0] / Mf;
    float var = (rq[0] - rs[0] * rs[0] / Mf) / (Mf - 1.0f);
    meanA[b] = mu;
    istdA[b] = rsqrtf(var);
  }
}

// ---------------- potentials via WMMA ----------------
// One wave handles a tile of 16 (b,i,j) cells x 32 labels.
// Lane l and lane l+16 hold the two 16-label halves of cell (tile*16 + (l&15)).

__global__ void k_pot(const float* __restrict__ lp, const float* __restrict__ ev,
                      const float* __restrict__ meanA, const float* __restrict__ istdA,
                      float* __restrict__ potF, float* __restrict__ potS,
                      float* __restrict__ potP, float* __restrict__ wlpA) {
  int lane = threadIdx.x & 31;
  int tile = blockIdx.x * 8 + (threadIdx.x >> 5);  // 65536 tiles total
  int r = lane & 15;
  int half = lane >> 4;
  int b = tile >> 12;                              // 4096 tiles per batch
  float mu = meanA[b], is = istdA[b];
  size_t base = ((size_t)tile * 16 + (size_t)r) * Ln + (size_t)half * 16;
  __builtin_prefetch(ev + base, 0, 0);

  float x[16], e[16];
  const float4* xp = (const float4*)(lp + base);
  const float4* ep = (const float4*)(ev + base);
#pragma unroll
  for (int q = 0; q < 4; ++q) {
    float4 a = xp[q]; float4 c = ep[q];
    x[4*q+0] = a.x; x[4*q+1] = a.y; x[4*q+2] = a.z; x[4*q+3] = a.w;
    e[4*q+0] = c.x; e[4*q+1] = c.y; e[4*q+2] = c.z; e[4*q+3] = c.w;
  }
#pragma unroll
  for (int k = 0; k < 16; ++k) x[k] = (x[k] - mu) * is;

  v16h ones;
#pragma unroll
  for (int k = 0; k < 16; ++k) ones[k] = (_Float16)1.0f;
  v8f c0 = {0.f, 0.f, 0.f, 0.f, 0.f, 0.f, 0.f, 0.f};

  // -------- full: logsumexp_l(x)  and  E-weighted sum for entropy --------
  float mF = x[0];
#pragma unroll
  for (int k = 1; k < 16; ++k) mF = fmaxf(mF, x[k]);
  mF = fmaxf(mF, __shfl_xor(mF, 16, 32));
  v16h aF, aW;
#pragma unroll
  for (int k = 0; k < 16; ++k) {
    float ex = __expf(x[k] - mF);
    aF[k] = (_Float16)ex;
    aW[k] = (_Float16)(ex * x[k]);
  }
  v8f cF = __builtin_amdgcn_wmma_f32_16x16x32_f16(false, aF, false, ones, (short)0, c0, false, false);
  v8f cW = __builtin_amdgcn_wmma_f32_16x16x32_f16(false, aW, false, ones, (short)0, c0, false, false);

  // -------- smooth: logsumexp_l(x + log(eval + 0.1*(1-eval) + 1e-10)) --------
  float y[16];
#pragma unroll
  for (int k = 0; k < 16; ++k) y[k] = x[k] + __logf(e[k] + (1.0f - e[k]) * 0.1f + 1e-10f);
  float mS = y[0];
#pragma unroll
  for (int k = 1; k < 16; ++k) mS = fmaxf(mS, y[k]);
  mS = fmaxf(mS, __shfl_xor(mS, 16, 32));
  v16h aS;
#pragma unroll
  for (int k = 0; k < 16; ++k) aS[k] = (_Float16)__expf(y[k] - mS);
  v8f cS = __builtin_amdgcn_wmma_f32_16x16x32_f16(false, aS, false, ones, (short)0, c0, false, false);

  // -------- partial: logsumexp_l(x - 1e6*(1-eval)) --------
#pragma unroll
  for (int k = 0; k < 16; ++k) y[k] = x[k] - 1.0e6f * (1.0f - e[k]);
  float mP = y[0];
#pragma unroll
  for (int k = 1; k < 16; ++k) mP = fmaxf(mP, y[k]);
  mP = fmaxf(mP, __shfl_xor(mP, 16, 32));
  v16h aP;
#pragma unroll
  for (int k = 0; k < 16; ++k) aP[k] = (_Float16)__expf(y[k] - mP);
  v8f cP = __builtin_amdgcn_wmma_f32_16x16x32_f16(false, aP, false, ones, (short)0, c0, false, false);

  float sF = gather_row(cF, lane);
  float sW = gather_row(cW, lane);
  float sS = gather_row(cS, lane);
  float sP = gather_row(cP, lane);
  if (lane < 16) {                       // lane == row == cell index in tile
    size_t o = (size_t)tile * 16 + lane;
    potF[o] = mF + __logf(sF);
    wlpA[o] = sW / sF;                   // E_l[x] under label softmax
    potS[o] = mS + __logf(sS);
    potP[o] = mP + __logf(sP);
  }
}

// ---------------- inside DP ----------------
// blockIdx = variant*16 + batch. cs (start,width) lives in padded LDS (257KB).
// ce (end,width) lives in global (L2-resident). 4 lanes cooperate per span.

__global__ void k_inside(const float* __restrict__ potBase, const int* __restrict__ lengths,
                         float* ceAll, float* csgAll, float* __restrict__ zout) {
  extern __shared__ float cs[];                       // [Nn][STR]
  int b = blockIdx.x & 15;
  int v = blockIdx.x >> 4;
  const float* pot = potBase + ((size_t)v * Bn + b) * Nn * Nn;   // pot[i*N + j]
  float* ceb = ceAll + ((size_t)v * Bn + b) * Nn * Nn;           // ceb[end*N + width]
  float* csb = csgAll + (size_t)b * Nn * Nn;                     // mirror (full only)
  int tid = threadIdx.x;

  if (tid < Nn) {                                     // width-0 diagonal
    float p = pot[tid * Nn + tid];
    cs[tid * STR] = p;
    ceb[(size_t)tid * Nn] = p;
    if (v == 0) csb[(size_t)tid * Nn] = p;
  }
  __threadfence_block();
  __syncthreads();

  int i = tid >> 2, sub = tid & 3;                    // 4 lanes per span
  for (int w = 1; w < Nn; ++w) {
    bool act = (i < Nn - w);
    float m = NEGF, s = 0.f;
    if (act) {
      const float* csrow = cs + i * STR;
      const float* cerow = ceb + (size_t)(i + w) * Nn;
      for (int u = sub; u < w; u += 4) {              // split at i+u
        float val = csrow[u] + cerow[w - 1 - u];      // beta(i,i+u)+beta(i+u+1,i+w)
        float mn = fmaxf(m, val);
        s = s * __expf(m - mn) + __expf(val - mn);
        m = mn;
      }
    }
    lse_merge(m, s, 1);
    lse_merge(m, s, 2);
    if (act && sub == 0) {
      float nv = m + __logf(s) + pot[i * Nn + (i + w)];
      cs[i * STR + w] = nv;
      ceb[(size_t)(i + w) * Nn + w] = nv;
      if (v == 0) csb[(size_t)i * Nn + w] = nv;
    }
    __threadfence_block();
    __syncthreads();
  }
  if (tid == 0) {
    int len = lengths[b];
    zout[v * Bn + b] = cs[len - 1];                   // beta(0, len-1)
  }
}

// ---------------- outside DP + entropy ----------------
// O(i,j) = LSE over parents; marg(i,j)=exp(O+beta-logZ); ent += marg * E[x].
// Spans not under the root (j >= len or width >= len) stay at NEG (unreachable).

__global__ void k_outside(const float* __restrict__ potF, const float* __restrict__ wl,
                          const float* __restrict__ cbAll, const float* __restrict__ zA,
                          const int* __restrict__ lengths, float* ogAll, float* __restrict__ entOut) {
  int b = blockIdx.x;
  int tid = threadIdx.x;
  const float* pot = potF + (size_t)b * Nn * Nn;   // pot[i*N + j]
  const float* wle = wl + (size_t)b * Nn * Nn;
  const float* cb  = cbAll + (size_t)b * Nn * Nn;  // beta in (start,width): cb[s*N + w]
  float* og = ogAll + (size_t)b * Nn * Nn;         // outside in (start,width)
  int len = lengths[b];
  float logZ = zA[b];

  for (int idx = tid; idx < Nn * Nn; idx += 1024) og[idx] = NEGF;
  __threadfence_block();
  __syncthreads();
  float ent = 0.f;
  if (tid == 0) {
    og[len - 1] = 0.f;            // root (0,len-1)
    ent = wle[len - 1];           // root marginal == 1
  }
  __threadfence_block();
  __syncthreads();

  int i = tid >> 2, sub = tid & 3;
  for (int v = Nn - 2; v >= 0; --v) {
    int j = i + v;
    bool act = (j < len) && !(i == 0 && v == len - 1);
    float m = NEGF, s = 0.f;
    if (act) {
      const float* ogi = og + (size_t)i * Nn;
      const float* poti = pot + (size_t)i * Nn;
      const float* betar = cb + (size_t)(j + 1) * Nn;   // beta(j+1, jp)
      for (int jp = j + 1 + sub; jp < len; jp += 4) {   // as left child of (i,jp)
        float t = ogi[jp - i] + poti[jp] + betar[jp - j - 1];
        float mn = fmaxf(m, t);
        s = s * __expf(m - mn) + __expf(t - mn);
        m = mn;
      }
      for (int h = sub; h < i; h += 4) {                // as right child of (h,j)
        float t = og[(size_t)h * Nn + (j - h)] + pot[(size_t)h * Nn + j] + cb[(size_t)h * Nn + (i - 1 - h)];
        float mn = fmaxf(m, t);
        s = s * __expf(m - mn) + __expf(t - mn);
        m = mn;
      }
    }
    lse_merge(m, s, 1);
    lse_merge(m, s, 2);
    if (act && sub == 0) {
      float O = m + __logf(fmaxf(s, 1e-38f));
      og[(size_t)i * Nn + v] = O;
      ent += __expf(O + cb[(size_t)i * Nn + v] - logZ) * wle[(size_t)i * Nn + j];
    }
    __threadfence_block();
    __syncthreads();
  }

  __shared__ float red[1024];
  red[tid] = ent;
  __syncthreads();
  for (int off = 512; off; off >>= 1) {
    if (tid < off) red[tid] += red[tid + off];
    __syncthreads();
  }
  if (tid == 0) entOut[b] = red[0];
}

// ---------------- finalize ----------------

__global__ void k_final(const float* __restrict__ zA, const float* __restrict__ entA,
                        float* __restrict__ out) {
  int t = threadIdx.x;
  if (t < Bn) {
    float zf = zA[t];            // full
    float zs = zA[Bn + t];       // smooth
    float zp = zA[2 * Bn + t];   // partial
    out[t]          = zp - zf;   // log_prob
    out[Bn + t]     = zs - zf;   // log_prob_smooth
    out[2 * Bn + t] = zf - entA[t];  // entropy
  }
}

// ---------------- launch ----------------

extern "C" void kernel_launch(void* const* d_in, const int* in_sizes, int n_in,
                              void* d_out, int out_size, void* d_ws, size_t ws_size,
                              hipStream_t stream) {
  (void)in_sizes; (void)n_in; (void)out_size; (void)ws_size;
  const float* lp      = (const float*)d_in[0];
  // d_in[1] (mask) is unused by the reference
  const int* lengths   = (const int*)d_in[2];
  const float* ev      = (const float*)d_in[3];
  float* out = (float*)d_out;
  float* ws  = (float*)d_ws;

  const size_t BNN = (size_t)Bn * Nn * Nn;
  float* meanA = ws;
  float* istdA = ws + 16;
  float* zA    = ws + 32;            // 3*B
  float* entA  = ws + 80;            // B
  float* partS = ws + 96;            // B*64
  float* partQ = ws + 96 + 1024;     // B*64
  float* big   = ws + 4096;
  float* potF  = big;                // B*N*N each
  float* potS  = potF + BNN;
  float* potP  = potS + BNN;
  float* wlpA  = potP + BNN;
  float* ceA   = wlpA + BNN;         // 3 * B*N*N  (end,width) charts
  float* csg   = ceA + 3 * BNN;      // full beta mirror (start,width)
  float* og    = csg + BNN;          // outside chart

  const int smem = Nn * STR * 4;     // 263168 B <= 320KB WGP LDS
  hipFuncSetAttribute((const void*)k_inside, hipFuncAttributeMaxDynamicSharedMemorySize, smem);

  k_stats_partial<<<Bn * 64, 256, 0, stream>>>(lp, partS, partQ);
  k_stats_final<<<Bn, 64, 0, stream>>>(partS, partQ, meanA, istdA);
  k_pot<<<(Bn * Nn * Nn / 16) / 8, 256, 0, stream>>>(lp, ev, meanA, istdA, potF, potS, potP, wlpA);
  k_inside<<<3 * Bn, 1024, smem, stream>>>(potF, lengths, ceA, csg, zA);
  k_outside<<<Bn, 1024, 0, stream>>>(potF, wlpA, csg, zA, lengths, og, entA);
  k_final<<<1, 32, 0, stream>>>(zA, entA, out);
}
